// Decoder_76974403879068
// MI455X (gfx1250) — matrix operand order repaired and checked by reading, not compiled, not run on hardware
//
#include <hip/hip_runtime.h>
#include <hip/hip_bf16.h>
#include <math.h>

typedef __attribute__((ext_vector_type(16))) _Float16 v16h;
typedef __attribute__((ext_vector_type(8)))  _Float16 v8h;
typedef __attribute__((ext_vector_type(8)))  float    v8f;
typedef unsigned int u32x4 __attribute__((ext_vector_type(4)));
typedef int          i32x8 __attribute__((ext_vector_type(8)));
typedef int          i32x4 __attribute__((ext_vector_type(4)));

#define HID    512
#define GATES  2048
#define OUTC   26
#define TLEN   96
#define BTILE  16
#define NBATCH 8192
#define XSTR   520            // halves per activation row (512 + 8 pad) -> 1040B, 16B aligned
#define WROW   72             // halves per staged weight row (64 + 8 pad) -> 144B stride (TDM pad reproduces this)
#define KC     64
#define NCHUNK 8
#define GSZ    (GATES * HID)

#define CFENCE() asm volatile("" ::: "memory")

__device__ __forceinline__ float sigmoidf_(float x) { return 1.0f / (1.0f + expf(-x)); }

// TDM: DMA one weight chunk (4 gates x 16 rows x 64 f16) global -> padded LDS rows.
// g1: data_size=2B(1<<16), pad_enable(1<<20), pad_interval=4 (32 dwords)<<22, pad_amount=3 (4 dwords)<<25,
//     tensor_dim0=64, tensor_dim1=16, tile_dim0=64, tile_dim1=16, tile_dim2=4,
//     tensor_dim0_stride=512, tensor_dim1_stride=512*512.
__device__ __forceinline__ void tdm_load_chunk(unsigned int lds_off, unsigned long long gaddr)
{
    u32x4 g0;
    g0.x = 1u;                                                   // count=1, user descriptor
    g0.y = lds_off;                                              // LDS byte address
    g0.z = (unsigned int)gaddr;                                  // global_addr[31:0]
    g0.w = ((unsigned int)(gaddr >> 32) & 0x01FFFFFFu) | 0x80000000u; // addr[56:32] | type=2
    i32x8 g1 = { (int)0x07110000u,      // mask=0, data_size=1, pad_en, interval=4, amount=3
                 (int)(64u << 16),      // tensor_dim0 = 64
                 (int)(16u << 16),      // tensor_dim1 = 16
                 (int)(64u << 16),      // tile_dim0 = 64
                 (int)(16u | (4u << 16)), // tile_dim1=16, tile_dim2=4
                 512,                   // tensor_dim0_stride low
                 0,                     // dim0_stride hi | dim1_stride low16
                 4 };                   // tensor_dim1_stride >> 16  (512*512 = 0x40000)
    i32x4 g2 = { 4, 0, 0, 0 };          // tensor_dim2 = 4
    i32x4 g3 = { 0, 0, 0, 0 };
    CFENCE();
#if __clang_major__ >= 23
    i32x8 g4 = { 0, 0, 0, 0, 0, 0, 0, 0 };
    __builtin_amdgcn_tensor_load_to_lds(g0, g1, g2, g3, g4, 0);
#else
    __builtin_amdgcn_tensor_load_to_lds(g0, g1, g2, g3, 0);
#endif
    CFENCE();
}

// One-time fp32 -> f16 weight conversion into workspace (halves steady-state L2 traffic).
__global__ void __launch_bounds__(256) cvt_weights_kernel(const float* __restrict__ wih,
                                                          const float* __restrict__ whh,
                                                          _Float16* __restrict__ out)
{
    const size_t n = (size_t)2 * GSZ;
    for (size_t i = (size_t)blockIdx.x * blockDim.x + threadIdx.x; i < n;
         i += (size_t)gridDim.x * blockDim.x) {
        out[i]     = (_Float16)wih[i];
        out[n + i] = (_Float16)whh[i];
    }
}

// Persistent decoder: one workgroup (8 wave32) owns 16 batch rows for all 96 steps.
__global__ void __launch_bounds__(256) lstm_decoder_kernel(
    const float* __restrict__ h0,      const float* __restrict__ c0,
    const float* __restrict__ act_emb, const float* __restrict__ mode_emb,
    const float* __restrict__ b_ih,    const float* __restrict__ b_hh,
    const float* __restrict__ fc_W,    const float* __restrict__ fc_b,
    const _Float16* __restrict__ wf16,
    float* __restrict__ out)
{
    __shared__ __attribute__((aligned(16))) _Float16 sh_x[BTILE * XSTR];
    __shared__ __attribute__((aligned(16))) _Float16 sh_h[2][2][BTILE * XSTR];
    __shared__ __attribute__((aligned(16))) _Float16 sh_w[8][2][64 * WROW];  // double-buffered TDM target
    __shared__ float    c_lds[2 * 8 * 4 * 8 * 32];      // [l][wave][t][r][lane], lane-major: conflict-free
    __shared__ float    predBuf[BTILE][OUTC];
    __shared__ float    nextin[BTILE][4];

    const int tid    = threadIdx.x;
    const int lane   = tid & 31;
    const int wid    = tid >> 5;
    const int hh     = lane >> 4;
    const int ln16   = lane & 15;
    const int batch0 = blockIdx.x * BTILE;

    const unsigned int ldsSlot[2] = {
        (unsigned int)(size_t)&sh_w[wid][0][0],
        (unsigned int)(size_t)&sh_w[wid][1][0]
    };

    // ---- initial state ----
    for (int i = tid; i < 2 * BTILE * HID; i += 256) {
        int l = i / (BTILE * HID);
        int r = i - l * BTILE * HID;
        int m = r / HID, k = r - m * HID;
        sh_h[l][0][m * XSTR + k] =
            (_Float16)h0[(size_t)l * NBATCH * HID + (size_t)(batch0 + m) * HID + k];
    }
    #pragma unroll
    for (int l = 0; l < 2; ++l)
        #pragma unroll
        for (int t = 0; t < 4; ++t)
            #pragma unroll
            for (int r = 0; r < 8; ++r) {
                int m   = r + hh * 8;
                int col = wid * 64 + t * 16 + ln16;
                c_lds[((((l * 8 + wid) * 4 + t) * 8) + r) * 32 + lane] =
                    c0[(size_t)l * NBATCH * HID + (size_t)(batch0 + m) * HID + col];
            }
    if (tid < BTILE) {  // SOS input: act=0, dur=0, mode=0, dist=0
        nextin[tid][0] = 0.f; nextin[tid][1] = 0.f; nextin[tid][2] = 0.f; nextin[tid][3] = 0.f;
    }
    __syncthreads();

    int p = 0;
    for (int step = 0; step < TLEN; ++step) {
        // ---- embed: x = [act_emb[a] | dur | mode_emb[m] | dist] ----
        for (int i = tid; i < BTILE * HID; i += 256) {
            int m = i >> 9, k = i & 511;
            float v;
            if (k < 382)       v = act_emb[(int)nextin[m][0] * 382 + k];
            else if (k == 382) v = nextin[m][1];
            else if (k < 511)  v = mode_emb[(int)nextin[m][2] * 128 + (k - 383)];
            else               v = nextin[m][3];
            sh_x[m * XSTR + k] = (_Float16)v;
        }
        __syncthreads();

        // ---- 2 LSTM layers ----
        for (int l = 0; l < 2; ++l) {
            const _Float16* Xs = (l == 0) ? sh_x : &sh_h[0][1 - p][0];
            const _Float16* Hs = &sh_h[l][p][0];
            _Float16*       Hd = &sh_h[l][1 - p][0];

            for (int t = 0; t < 4; ++t) {
                const int ncol = wid * 64 + t * 16 + ln16;
                v8f acc[4];
                #pragma unroll
                for (int g = 0; g < 4; ++g) {
                    float b = b_ih[l * GATES + g * 512 + ncol] + b_hh[l * GATES + g * 512 + ncol];
                    v8f tmp = {b, b, b, b, b, b, b, b};
                    acc[g] = tmp;
                }
                #pragma unroll
                for (int src = 0; src < 2; ++src) {
                    const _Float16* X = src ? Hs : Xs;
                    const _Float16* W = wf16 + (size_t)src * (2 * (size_t)GSZ) + (size_t)l * GSZ;
                    const unsigned long long wbase =
                        (unsigned long long)(size_t)(W + (size_t)(wid * 64 + t * 16) * HID);

                    tdm_load_chunk(ldsSlot[0], wbase);           // prime chunk 0
                    for (int chunk = 0; chunk < NCHUNK; ++chunk) {
                        if (chunk + 1 < NCHUNK) {                // DMA next chunk into other slot
                            tdm_load_chunk(ldsSlot[(chunk + 1) & 1],
                                           wbase + (unsigned long long)((chunk + 1) * KC) * 2ull);
                            __builtin_amdgcn_s_wait_tensorcnt(1);  // current chunk's DMA done
                        } else {
                            __builtin_amdgcn_s_wait_tensorcnt(0);  // last chunk: all DMAs done
                        }
                        CFENCE();
                        const _Float16* ws = &sh_w[wid][chunk & 1][0];
                        const int cb = chunk * KC;
                        #pragma unroll
                        for (int ks = 0; ks < 2; ++ks) {
                            const int kb = cb + ks * 32;
                            // A fragment: 16-bit A 16x32 layout (lane m, K runs of 8)
                            union { v16h v; v8h h[2]; } ua;
                            ua.h[0] = *(const v8h*)(X + ln16 * XSTR + kb + hh * 8);
                            ua.h[1] = *(const v8h*)(X + ln16 * XSTR + kb + 16 + hh * 8);
                            #pragma unroll
                            for (int g = 0; g < 4; ++g) {
                                // B fragment: 16-bit B 32x16 layout (lane n, K runs of 8)
                                const _Float16* bp = ws + (g * 16 + ln16) * WROW + ks * 32 + hh * 8;
                                union { v16h v; v8h h[2]; } ub;
                                ub.h[0] = *(const v8h*)bp;
                                ub.h[1] = *(const v8h*)(bp + 16);
                                acc[g] = __builtin_amdgcn_wmma_f32_16x16x32_f16(
                                    false, ua.v, false, ub.v, (short)0, acc[g], false, false);
                            }
                        }
                        CFENCE();
                    }
                }
                // elementwise LSTM (wave-local; c in lane-major LDS)
                float* cp = &c_lds[(((l * 8 + wid) * 4 + t) * 8) * 32 + lane];
                #pragma unroll
                for (int r = 0; r < 8; ++r) {
                    float iv = sigmoidf_(acc[0][r]);
                    float fv = sigmoidf_(acc[1][r]);
                    float gv = tanhf(acc[2][r]);
                    float ov = sigmoidf_(acc[3][r]);
                    float cn = fv * cp[r * 32] + iv * gv;
                    float hn = ov * tanhf(cn);
                    cp[r * 32] = cn;
                    int m = r + hh * 8;
                    Hd[m * XSTR + ncol] = (_Float16)hn;
                }
            }
            __syncthreads();
        }
        p ^= 1;

        // ---- FC head (26 outputs/row): 416 tasks over 256 threads ----
        for (int task = tid; task < BTILE * OUTC; task += 256) {
            int row = task / OUTC, oc = task - row * OUTC;
            const _Float16* hrow = &sh_h[1][p][row * XSTR];
            const float* w = fc_W + oc * HID;
            float dot = fc_b[oc];
            for (int k = 0; k < HID; ++k) dot += w[k] * (float)hrow[k];
            predBuf[row][oc] = dot;
            out[((size_t)(batch0 + row) * TLEN + step) * OUTC + oc] = dot;
        }
        __syncthreads();
        // ---- argmax feedback (first-max semantics like jnp.argmax) ----
        if (tid < BTILE) {
            float best = predBuf[tid][0]; int bi = 0;
            for (int j = 1; j < 16; ++j) { float v = predBuf[tid][j]; if (v > best) { best = v; bi = j; } }
            float bm = predBuf[tid][17]; int mi = 0;
            for (int j = 1; j < 8; ++j) { float v = predBuf[tid][17 + j]; if (v > bm) { bm = v; mi = j; } }
            nextin[tid][0] = (float)bi;
            nextin[tid][1] = 1.0f;
            nextin[tid][2] = (float)mi;
            nextin[tid][3] = sigmoidf_(predBuf[tid][25]);
        }
        __syncthreads();
    }

    // ---- output transforms (all local to this WG's batch rows) ----
    __threadfence_block();
    __syncthreads();
    for (int i = tid; i < BTILE * TLEN; i += 256) {
        int b = i / TLEN, t = i - b * TLEN;
        float* o = out + ((size_t)(batch0 + b) * TLEN + t) * OUTC;
        float mx = o[0];
        for (int j = 1; j < 16; ++j) mx = fmaxf(mx, o[j]);
        float s = 0.f;
        for (int j = 0; j < 16; ++j) s += expf(o[j] - mx);
        float ls = logf(s);
        for (int j = 0; j < 16; ++j) o[j] = o[j] - mx - ls;      // log_softmax acts
        float mx2 = o[17];
        for (int j = 18; j < 25; ++j) mx2 = fmaxf(mx2, o[j]);
        float s2 = 0.f;
        for (int j = 17; j < 25; ++j) s2 += expf(o[j] - mx2);
        float ls2 = logf(s2);
        for (int j = 17; j < 25; ++j) o[j] = o[j] - mx2 - ls2;   // log_softmax modes
        o[25] = sigmoidf_(o[25]);                                 // sigmoid dist
    }
    if (tid < BTILE) {   // softmax of durations over time (axis=-2), col 16 only: disjoint
        int b = tid;
        float mx = -INFINITY;
        for (int t = 0; t < TLEN; ++t)
            mx = fmaxf(mx, out[((size_t)(batch0 + b) * TLEN + t) * OUTC + 16]);
        float s = 0.f;
        for (int t = 0; t < TLEN; ++t)
            s += expf(out[((size_t)(batch0 + b) * TLEN + t) * OUTC + 16] - mx);
        float inv = 1.0f / s;
        for (int t = 0; t < TLEN; ++t) {
            size_t ix = ((size_t)(batch0 + b) * TLEN + t) * OUTC + 16;
            out[ix] = expf(out[ix] - mx) * inv;
        }
    }
}

extern "C" void kernel_launch(void* const* d_in, const int* in_sizes, int n_in,
                              void* d_out, int out_size, void* d_ws, size_t ws_size,
                              hipStream_t stream)
{
    (void)in_sizes; (void)n_in; (void)out_size; (void)ws_size;
    const float* h0       = (const float*)d_in[1];
    const float* c0       = (const float*)d_in[2];
    const float* act_emb  = (const float*)d_in[3];
    const float* mode_emb = (const float*)d_in[4];
    const float* W_ih     = (const float*)d_in[5];
    const float* W_hh     = (const float*)d_in[6];
    const float* b_ih     = (const float*)d_in[7];
    const float* b_hh     = (const float*)d_in[8];
    const float* fc_W     = (const float*)d_in[9];
    const float* fc_b     = (const float*)d_in[10];
    _Float16* wf16 = (_Float16*)d_ws;       // needs 4 * 2048 * 512 * 2B = 8 MiB of scratch
    float* outp = (float*)d_out;

    cvt_weights_kernel<<<2048, 256, 0, stream>>>(W_ih, W_hh, wf16);
    lstm_decoder_kernel<<<NBATCH / BTILE, 256, 0, stream>>>(
        h0, c0, act_emb, mode_emb, b_ih, b_hh, fc_W, fc_b, wf16, outp);
}
